// GNN_2_Model_57080115363951
// MI455X (gfx1250) — matrix operand order repaired and checked by reading, not compiled, not run on hardware
//
#include <hip/hip_runtime.h>
#include <hip/hip_bf16.h>

// MI455X (gfx1250, wave32) 2-layer GAT + heads.
// - GEMMs: v_wmma_f32_16x16x32_f16, B panels staged to LDS with
//   global_load_async_to_lds_b128 (+ s_wait_asynccnt), A via b128 global loads.
// - Edge softmax/aggregation: f16 gathers + hardware f32 global atomics.
// Workspace requirement: ~404 MB.

typedef __attribute__((ext_vector_type(16))) _Float16 v16h;
typedef __attribute__((ext_vector_type(8)))  _Float16 v8h;
typedef __attribute__((ext_vector_type(8)))  float    v8f;

#define NN 50000
#define NE 500000
#define NP 131072
#define DLM 768
#define DMF 64
#define DIN 832
#define DOUT 512

// ---------------------------------------------------------------- helpers ---
__device__ __forceinline__ unsigned f2ord(float x) {
    unsigned u = __float_as_uint(x);
    return (u & 0x80000000u) ? ~u : (u | 0x80000000u);
}
__device__ __forceinline__ float ord2f(unsigned u) {
    unsigned b = (u & 0x80000000u) ? (u & 0x7FFFFFFFu) : ~u;
    return __uint_as_float(b);
}
__device__ __forceinline__ void atomic_add_f32(float* p, float v) {
    __hip_atomic_fetch_add(p, v, __ATOMIC_RELAXED, __HIP_MEMORY_SCOPE_AGENT);
}

// Async copy 16B/lane from global to LDS (gfx1250), tracked by ASYNCcnt.
__device__ __forceinline__ void async_ldsB(const _Float16* src, unsigned lds_byte_addr) {
    asm volatile("global_load_async_to_lds_b128 %0, %1, off"
                 :: "v"(lds_byte_addr), "v"(src) : "memory");
}
__device__ __forceinline__ void wait_async0() {
    asm volatile("s_wait_asynccnt 0x0" ::: "memory");
}

// A fragment for V_WMMA_F32_16X16X32_F16 (ISA 7.12.2, 16-bit A 16x32):
// lane m = lane&15; halves 0..7 = K kb..kb+7, halves 8..15 = K kb+16..kb+23,
// kb += 8 for lanes 16..31.  Two contiguous 16B loads per lane.
__device__ __forceinline__ v16h load_a_frag(const _Float16* __restrict__ rowA,
                                            int kt, int lane) {
    const int kb = kt * 32 + ((lane & 16) ? 8 : 0);
    v8h lo = *(const v8h*)(rowA + kb);
    v8h hi = *(const v8h*)(rowA + kb + 16);
    v16h a;
#pragma unroll
    for (int i = 0; i < 8; ++i) { a[i] = lo[i]; a[8 + i] = hi[i]; }
    return a;
}

// ------------------------------------------------------------ weight pack ---
// Pack row-major f32 W[K,Ncols] into per-tile B-fragment order:
// out[((kt*nTiles+nt)*32+lane)*16+i] = W[kt*32 + (lane>=16?8:0) + (i<8?i:i+8)][nt*16 + (lane&15)]
__global__ void pack_w_kernel(const float* __restrict__ W, _Float16* __restrict__ out,
                              int K, int Ncols) {
    const int tid = blockIdx.x * blockDim.x + threadIdx.x;
    const int nTiles = Ncols >> 4;
    const int kTiles = K >> 5;
    if (tid >= kTiles * nTiles * 32) return;
    const int lane = tid & 31;
    const int tile = tid >> 5;
    const int nt = tile % nTiles;
    const int kt = tile / nTiles;
    const int col = nt * 16 + (lane & 15);
    const int kb  = kt * 32 + ((lane & 16) ? 8 : 0);
#pragma unroll
    for (int i = 0; i < 16; ++i)
        out[(size_t)tid * 16 + i] = (_Float16)W[(size_t)(kb + i + (i >= 8 ? 8 : 0)) * Ncols + col];
}

// ------------------------------------------------------- embedding fusion ---
__global__ void fuse_embed_kernel(const float* __restrict__ cell_h,
                                  const float* __restrict__ pos_emb,
                                  const float* __restrict__ mf_emb,
                                  const int* __restrict__ row_pos,
                                  const int* __restrict__ col_pos,
                                  const int* __restrict__ row_mf,
                                  const int* __restrict__ col_mf,
                                  _Float16* __restrict__ out) {
    const size_t tid = (size_t)blockIdx.x * blockDim.x + threadIdx.x;
    if (tid >= (size_t)NN * DIN) return;
    const int n = (int)(tid / DIN);
    const int c = (int)(tid % DIN);
    float v;
    if (c < DLM) {
        v = cell_h[(size_t)n * DLM + c]
          + pos_emb[(size_t)row_pos[n] * DLM + c]
          + pos_emb[(size_t)col_pos[n] * DLM + c];
    } else {
        const int cc = c - DLM;
        const int rm = row_mf[n] < 5 ? row_mf[n] : 5;
        const int cm = col_mf[n] < 5 ? col_mf[n] : 5;
        v = mf_emb[rm * DMF + cc] + mf_emb[cm * DMF + cc];
    }
    out[tid] = (_Float16)v;
}

// ------------------------------------------------------------- WMMA GEMMs ---
// Block = 8 waves sharing one 64-column group; B staged to LDS in 8-ktile
// (32 KB) phases via async copy; each wave owns one 16-row m-tile.

#define PHASE_KT 8

// feat = A[M,K] @ Bp -> f16 C[M,Ncols]
__global__ void gemm_feat_kernel(const _Float16* __restrict__ A,
                                 const _Float16* __restrict__ Bp,
                                 _Float16* __restrict__ C,
                                 int M, int K, int Ncols) {
    __shared__ _Float16 ldsB[PHASE_KT * 2048];
    const int tid = threadIdx.x;
    const int lane = tid & 31;
    const int waveId = tid >> 5;
    const int nGroups = Ncols >> 6;
    const int nTiles = Ncols >> 4;
    const int nGroup = blockIdx.x % nGroups;
    const int mTile = (blockIdx.x / nGroups) * 8 + waveId;
    const bool valid = (mTile * 16) < M;
    const _Float16* rowA = A + (size_t)((valid ? mTile * 16 : 0) + (lane & 15)) * K;
    const unsigned lbase = (unsigned)(uintptr_t)ldsB;
    v8f acc[4] = {{}, {}, {}, {}};
    const int kTiles = K >> 5;
    for (int kt0 = 0; kt0 < kTiles; kt0 += PHASE_KT) {
        const int kc = (kTiles - kt0) < PHASE_KT ? (kTiles - kt0) : PHASE_KT;
        __syncthreads();
        for (int k = 0; k < kc; ++k)
            async_ldsB(Bp + ((size_t)(kt0 + k) * nTiles + nGroup * 4) * 512 + tid * 8,
                       lbase + k * 4096 + tid * 16);
        wait_async0();
        __syncthreads();
        for (int k = 0; k < kc; ++k) {
            const v16h a = load_a_frag(rowA, kt0 + k, lane);
            const _Float16* bl = ldsB + k * 2048 + lane * 16;
#pragma unroll
            for (int j = 0; j < 4; ++j) {
                const v16h b = *(const v16h*)(bl + j * 512);
                acc[j] = __builtin_amdgcn_wmma_f32_16x16x32_f16(false, a, false, b, (short)0, acc[j], false, false);
            }
        }
    }
    if (!valid) return;
    const int rowBase = mTile * 16 + ((lane & 16) ? 8 : 0);
    const int col = lane & 15;
#pragma unroll
    for (int j = 0; j < 4; ++j) {
        const int n0 = nGroup * 64 + j * 16 + col;
#pragma unroll
        for (int r = 0; r < 8; ++r)
            C[(size_t)(rowBase + r) * Ncols + n0] = (_Float16)acc[j][r];
    }
}

// z = relu(BN(A@Bp + bias)) -> f16 Z[M,Ncols]
__global__ void gemm_bn_kernel(const _Float16* __restrict__ A,
                               const _Float16* __restrict__ Bp,
                               const float* __restrict__ bias,
                               const float* __restrict__ gamma,
                               const float* __restrict__ beta,
                               const float* __restrict__ mean,
                               const float* __restrict__ var,
                               _Float16* __restrict__ Z,
                               int M, int K, int Ncols) {
    __shared__ _Float16 ldsB[PHASE_KT * 2048];
    const int tid = threadIdx.x;
    const int lane = tid & 31;
    const int waveId = tid >> 5;
    const int nGroups = Ncols >> 6;
    const int nTiles = Ncols >> 4;
    const int nGroup = blockIdx.x % nGroups;
    const int mTile = (blockIdx.x / nGroups) * 8 + waveId;
    const bool valid = (mTile * 16) < M;
    const _Float16* rowA = A + (size_t)((valid ? mTile * 16 : 0) + (lane & 15)) * K;
    const unsigned lbase = (unsigned)(uintptr_t)ldsB;
    v8f acc[4] = {{}, {}, {}, {}};
    const int kTiles = K >> 5;
    for (int kt0 = 0; kt0 < kTiles; kt0 += PHASE_KT) {
        const int kc = (kTiles - kt0) < PHASE_KT ? (kTiles - kt0) : PHASE_KT;
        __syncthreads();
        for (int k = 0; k < kc; ++k)
            async_ldsB(Bp + ((size_t)(kt0 + k) * nTiles + nGroup * 4) * 512 + tid * 8,
                       lbase + k * 4096 + tid * 16);
        wait_async0();
        __syncthreads();
        for (int k = 0; k < kc; ++k) {
            const v16h a = load_a_frag(rowA, kt0 + k, lane);
            const _Float16* bl = ldsB + k * 2048 + lane * 16;
#pragma unroll
            for (int j = 0; j < 4; ++j) {
                const v16h b = *(const v16h*)(bl + j * 512);
                acc[j] = __builtin_amdgcn_wmma_f32_16x16x32_f16(false, a, false, b, (short)0, acc[j], false, false);
            }
        }
    }
    if (!valid) return;
    const int rowBase = mTile * 16 + ((lane & 16) ? 8 : 0);
    const int col = lane & 15;
#pragma unroll
    for (int j = 0; j < 4; ++j) {
        const int n0 = nGroup * 64 + j * 16 + col;
        const float bb = bias[n0], mu = mean[n0];
        const float sc = rsqrtf(var[n0] + 1e-5f) * gamma[n0];
        const float bt = beta[n0];
#pragma unroll
        for (int r = 0; r < 8; ++r) {
            float x = (acc[j][r] + bb - mu) * sc + bt;
            x = x > 0.0f ? x : 0.0f;
            Z[(size_t)(rowBase + r) * Ncols + n0] = (_Float16)x;
        }
    }
}

// Edge head: A rows gathered as concat(h[pairs[p,0]], h[pairs[p,1]]); K=1024.
__global__ void gemm_edge_bn_kernel(const _Float16* __restrict__ H,
                                    const int* __restrict__ pairs,
                                    const _Float16* __restrict__ Bp,
                                    const float* __restrict__ bias,
                                    const float* __restrict__ gamma,
                                    const float* __restrict__ beta,
                                    const float* __restrict__ mean,
                                    const float* __restrict__ var,
                                    _Float16* __restrict__ Z,
                                    int Ncols) {
    __shared__ _Float16 ldsB[PHASE_KT * 2048];
    const int tid = threadIdx.x;
    const int lane = tid & 31;
    const int waveId = tid >> 5;
    const int nGroups = Ncols >> 6;
    const int nTiles = Ncols >> 4;
    const int nGroup = blockIdx.x % nGroups;
    const int mTile = (blockIdx.x / nGroups) * 8 + waveId;
    const bool valid = (mTile * 16) < NP;
    const int p = (valid ? mTile * 16 : 0) + (lane & 15);
    const _Float16* r0 = H + (size_t)pairs[p * 2 + 0] * DOUT;
    const _Float16* r1 = H + (size_t)pairs[p * 2 + 1] * DOUT;
    const unsigned lbase = (unsigned)(uintptr_t)ldsB;
    v8f acc[4] = {{}, {}, {}, {}};
    for (int kt0 = 0; kt0 < 32; kt0 += PHASE_KT) {   // K = 1024 -> 32 ktiles
        __syncthreads();
        for (int k = 0; k < PHASE_KT; ++k)
            async_ldsB(Bp + ((size_t)(kt0 + k) * nTiles + nGroup * 4) * 512 + tid * 8,
                       lbase + k * 4096 + tid * 16);
        wait_async0();
        __syncthreads();
        for (int k = 0; k < PHASE_KT; ++k) {
            const int kt = kt0 + k;
            const _Float16* rowA = (kt < 16) ? r0 : r1;
            const v16h a = load_a_frag(rowA, kt & 15, lane);
            const _Float16* bl = ldsB + k * 2048 + lane * 16;
#pragma unroll
            for (int j = 0; j < 4; ++j) {
                const v16h b = *(const v16h*)(bl + j * 512);
                acc[j] = __builtin_amdgcn_wmma_f32_16x16x32_f16(false, a, false, b, (short)0, acc[j], false, false);
            }
        }
    }
    if (!valid) return;
    const int rowBase = mTile * 16 + ((lane & 16) ? 8 : 0);
    const int col = lane & 15;
#pragma unroll
    for (int j = 0; j < 4; ++j) {
        const int n0 = nGroup * 64 + j * 16 + col;
        const float bb = bias[n0], mu = mean[n0];
        const float sc = rsqrtf(var[n0] + 1e-5f) * gamma[n0];
        const float bt = beta[n0];
#pragma unroll
        for (int r = 0; r < 8; ++r) {
            float x = (acc[j][r] + bb - mu) * sc + bt;
            x = x > 0.0f ? x : 0.0f;
            Z[(size_t)(rowBase + r) * Ncols + n0] = (_Float16)x;
        }
    }
}

// --------------------------------------------------------- GAT attention ---
__global__ void elr_kernel(const _Float16* __restrict__ feat,
                           const float* __restrict__ al, const float* __restrict__ ar,
                           float* __restrict__ el, float* __restrict__ er) {
    const int tid = blockIdx.x * blockDim.x + threadIdx.x;
    if (tid >= NN * 4) return;
    const int n = tid >> 2, h = tid & 3;
    const _Float16* f = feat + (size_t)n * DOUT + h * 128;
    const float* a = al + h * 128;
    const float* b = ar + h * 128;
    float sl = 0.0f, sr = 0.0f;
    for (int d = 0; d < 128; ++d) {
        const float fv = (float)f[d];
        sl += fv * a[d]; sr += fv * b[d];
    }
    el[tid] = sl; er[tid] = sr;
}

__global__ void edge_max_kernel(const int* __restrict__ src, const int* __restrict__ dst,
                                const float* __restrict__ el, const float* __restrict__ er,
                                unsigned* __restrict__ emax) {
    const int e = blockIdx.x * blockDim.x + threadIdx.x;
    if (e >= NE) return;
    const int s = src[e], d = dst[e];
#pragma unroll
    for (int h = 0; h < 4; ++h) {
        float x = el[s * 4 + h] + er[d * 4 + h];
        x = x > 0.0f ? x : 0.2f * x;
        atomicMax(&emax[d * 4 + h], f2ord(x));
    }
}

__global__ void edge_exp_kernel(const int* __restrict__ src, const int* __restrict__ dst,
                                const float* __restrict__ el, const float* __restrict__ er,
                                const unsigned* __restrict__ emax,
                                float* __restrict__ ee, float* __restrict__ denom) {
    const int e = blockIdx.x * blockDim.x + threadIdx.x;
    if (e >= NE) return;
    const int s = src[e], d = dst[e];
#pragma unroll
    for (int h = 0; h < 4; ++h) {
        float x = el[s * 4 + h] + er[d * 4 + h];
        x = x > 0.0f ? x : 0.2f * x;
        const float v = __expf(x - ord2f(emax[d * 4 + h]));
        ee[(size_t)e * 4 + h] = v;
        atomic_add_f32(&denom[d * 4 + h], v);
    }
}

__global__ void edge_agg_kernel(const int* __restrict__ src, const int* __restrict__ dst,
                                const float* __restrict__ ee, const float* __restrict__ denom,
                                const _Float16* __restrict__ feat, float* __restrict__ agg) {
    const size_t tid = (size_t)blockIdx.x * blockDim.x + threadIdx.x;
    if (tid >= (size_t)NE * 128) return;
    const int e = (int)(tid >> 7);
    const int dd = (int)(tid & 127);
    const int s = src[e], d = dst[e];
#pragma unroll
    for (int h = 0; h < 4; ++h) {
        const float alpha = ee[(size_t)e * 4 + h] / denom[d * 4 + h];
        atomic_add_f32(&agg[(size_t)d * DOUT + h * 128 + dd],
                       (float)feat[(size_t)s * DOUT + h * 128 + dd] * alpha);
    }
}

// out = agg + bias (+ residual); write f16 (and optionally f32)
__global__ void bias_res_kernel(const float* __restrict__ agg,
                                const float* __restrict__ bias,
                                const float* __restrict__ res,   // nullable
                                float* __restrict__ outf32,      // nullable
                                _Float16* __restrict__ outf16) {
    const size_t tid = (size_t)blockIdx.x * blockDim.x + threadIdx.x;
    if (tid >= (size_t)NN * DOUT) return;
    const int c = (int)(tid % DOUT);
    float v = agg[tid] + bias[c];
    if (res) v += res[tid];
    if (outf32) outf32[tid] = v;
    outf16[tid] = (_Float16)v;
}

// ------------------------------------------------------------ tiny heads ---
__global__ void gid_out_kernel(const _Float16* __restrict__ z,
                               const float* __restrict__ Wg2, const float* __restrict__ bg2,
                               float* __restrict__ out) {
    const int tid = blockIdx.x * blockDim.x + threadIdx.x;
    if (tid >= NN * 4) return;
    const int n = tid >> 2, j = tid & 3;
    const _Float16* zr = z + (size_t)n * 256;
    float s = bg2[j];
    for (int c = 0; c < 256; ++c) s += (float)zr[c] * Wg2[c * 4 + j];
    out[tid] = s;
}

__global__ void edge_out_kernel(const _Float16* __restrict__ z,
                                const float* __restrict__ We2, const float* __restrict__ be2,
                                float* __restrict__ out) {
    const int p = blockIdx.x * blockDim.x + threadIdx.x;
    if (p >= NP) return;
    const _Float16* zr = z + (size_t)p * 256;
    float s = be2[0];
    for (int c = 0; c < 256; ++c) s += (float)zr[c] * We2[c];
    out[(size_t)NN * 4 + p] = s;
}

// ---------------------------------------------------------------- launch ---
extern "C" void kernel_launch(void* const* d_in, const int* in_sizes, int n_in,
                              void* d_out, int out_size, void* d_ws, size_t ws_size,
                              hipStream_t stream) {
    (void)in_sizes; (void)n_in; (void)out_size; (void)ws_size;
    const float* cell_h  = (const float*)d_in[0];
    const float* pos_emb = (const float*)d_in[1];
    const float* mf_emb  = (const float*)d_in[2];
    const float* W1  = (const float*)d_in[3];
    const float* al1 = (const float*)d_in[4];
    const float* ar1 = (const float*)d_in[5];
    const float* b1  = (const float*)d_in[6];
    const float* W2  = (const float*)d_in[7];
    const float* al2 = (const float*)d_in[8];
    const float* ar2 = (const float*)d_in[9];
    const float* b2  = (const float*)d_in[10];
    const float* Wg1 = (const float*)d_in[11];
    const float* bg1 = (const float*)d_in[12];
    const float* g_gamma = (const float*)d_in[13];
    const float* g_beta  = (const float*)d_in[14];
    const float* g_mean  = (const float*)d_in[15];
    const float* g_var   = (const float*)d_in[16];
    const float* Wg2 = (const float*)d_in[17];
    const float* bg2 = (const float*)d_in[18];
    const float* We1 = (const float*)d_in[19];
    const float* be1 = (const float*)d_in[20];
    const float* e_gamma = (const float*)d_in[21];
    const float* e_beta  = (const float*)d_in[22];
    const float* e_mean  = (const float*)d_in[23];
    const float* e_var   = (const float*)d_in[24];
    const float* We2 = (const float*)d_in[25];
    const float* be2 = (const float*)d_in[26];
    const int* src = (const int*)d_in[27];
    const int* dst = (const int*)d_in[28];
    const int* row_pos = (const int*)d_in[29];
    const int* col_pos = (const int*)d_in[30];
    const int* row_mf  = (const int*)d_in[31];
    const int* col_mf  = (const int*)d_in[32];
    const int* pairs   = (const int*)d_in[33];
    float* out = (float*)d_out;

    // ---- workspace layout (bytes, 256-aligned; overlays noted) ----
    char* ws = (char*)d_ws;
    const size_t o_w1p  = 0;                               //   851,968
    const size_t o_w2p  = o_w1p  + 851968;                 //   524,288
    const size_t o_wg1p = o_w2p  + 524288;                 //   262,144
    const size_t o_we1p = o_wg1p + 262144;                 //   524,288
    const size_t o_h016 = o_we1p + 524288;                 // N*832 f16 ; reused h2f16
    const size_t o_feat = o_h016 + (size_t)NN * DIN * 2;   // N*512 f16
    const size_t o_agg  = o_feat + (size_t)NN * DOUT * 2;  // N*512 f32 ; reused ze16
    const size_t o_h1f32= o_agg  + (size_t)NN * DOUT * 4;  // N*512 f32 ; reused zg16
    const size_t o_h1f16= o_h1f32+ (size_t)NN * DOUT * 4;  // N*512 f16
    const size_t o_el   = o_h1f16+ (size_t)NN * DOUT * 2;
    const size_t o_er   = o_el   + (size_t)NN * 4 * 4;
    const size_t o_emax = o_er   + (size_t)NN * 4 * 4;
    const size_t o_den  = o_emax + (size_t)NN * 4 * 4;
    const size_t o_ee   = o_den  + (size_t)NN * 4 * 4;     // E*4 f32

    _Float16* w1p  = (_Float16*)(ws + o_w1p);
    _Float16* w2p  = (_Float16*)(ws + o_w2p);
    _Float16* wg1p = (_Float16*)(ws + o_wg1p);
    _Float16* we1p = (_Float16*)(ws + o_we1p);
    _Float16* h0f16 = (_Float16*)(ws + o_h016);
    _Float16* h2f16 = (_Float16*)(ws + o_h016);   // overlay (h0 dead after feat1)
    _Float16* feat  = (_Float16*)(ws + o_feat);
    float*    agg   = (float*)(ws + o_agg);
    _Float16* ze16  = (_Float16*)(ws + o_agg);    // overlay (agg dead after h2)
    float*    h1f32 = (float*)(ws + o_h1f32);
    _Float16* zg16  = (_Float16*)(ws + o_h1f32);  // overlay (h1f32 dead after h2)
    _Float16* h1f16 = (_Float16*)(ws + o_h1f16);
    float*    el    = (float*)(ws + o_el);
    float*    er    = (float*)(ws + o_er);
    unsigned* emax  = (unsigned*)(ws + o_emax);
    float*    denom = (float*)(ws + o_den);
    float*    ee    = (float*)(ws + o_ee);

    const int T = 256;
    // pack weights to WMMA B-fragment order
    pack_w_kernel<<<(26 * 32 * 32 + T - 1) / T, T, 0, stream>>>(W1, w1p, DIN, DOUT);
    pack_w_kernel<<<(16 * 32 * 32 + T - 1) / T, T, 0, stream>>>(W2, w2p, DOUT, DOUT);
    pack_w_kernel<<<(16 * 16 * 32 + T - 1) / T, T, 0, stream>>>(Wg1, wg1p, DOUT, 256);
    pack_w_kernel<<<(32 * 16 * 32 + T - 1) / T, T, 0, stream>>>(We1, we1p, 2 * DOUT, 256);

    // embedding fusion -> f16
    fuse_embed_kernel<<<(int)(((size_t)NN * DIN + T - 1) / T), T, 0, stream>>>(
        cell_h, pos_emb, mf_emb, row_pos, col_pos, row_mf, col_mf, h0f16);

    const int mB = (NN / 16 + 7) / 8;   // 391 m-blocks (8 waves each)

    // ---- GAT layer 1 ----
    gemm_feat_kernel<<<mB * (DOUT / 64), T, 0, stream>>>(h0f16, w1p, feat, NN, DIN, DOUT);
    elr_kernel<<<(NN * 4 + T - 1) / T, T, 0, stream>>>(feat, al1, ar1, el, er);
    hipMemsetAsync(emax, 0, (size_t)NN * 4 * 4 * 2, stream);   // emax + denom
    hipMemsetAsync(agg, 0, (size_t)NN * DOUT * 4, stream);
    edge_max_kernel<<<(NE + T - 1) / T, T, 0, stream>>>(src, dst, el, er, emax);
    edge_exp_kernel<<<(NE + T - 1) / T, T, 0, stream>>>(src, dst, el, er, emax, ee, denom);
    edge_agg_kernel<<<(int)(((size_t)NE * 128 + T - 1) / T), T, 0, stream>>>(src, dst, ee, denom, feat, agg);
    bias_res_kernel<<<(int)(((size_t)NN * DOUT + T - 1) / T), T, 0, stream>>>(agg, b1, nullptr, h1f32, h1f16);

    // ---- GAT layer 2 (identity residual) ----
    gemm_feat_kernel<<<mB * (DOUT / 64), T, 0, stream>>>(h1f16, w2p, feat, NN, DOUT, DOUT);
    elr_kernel<<<(NN * 4 + T - 1) / T, T, 0, stream>>>(feat, al2, ar2, el, er);
    hipMemsetAsync(emax, 0, (size_t)NN * 4 * 4 * 2, stream);
    hipMemsetAsync(agg, 0, (size_t)NN * DOUT * 4, stream);
    edge_max_kernel<<<(NE + T - 1) / T, T, 0, stream>>>(src, dst, el, er, emax);
    edge_exp_kernel<<<(NE + T - 1) / T, T, 0, stream>>>(src, dst, el, er, emax, ee, denom);
    edge_agg_kernel<<<(int)(((size_t)NE * 128 + T - 1) / T), T, 0, stream>>>(src, dst, ee, denom, feat, agg);
    bias_res_kernel<<<(int)(((size_t)NN * DOUT + T - 1) / T), T, 0, stream>>>(agg, b2, h1f32, nullptr, h2f16);

    // ---- comp_and_gid head ----
    gemm_bn_kernel<<<mB * (256 / 64), T, 0, stream>>>(
        h2f16, wg1p, bg1, g_gamma, g_beta, g_mean, g_var, zg16, NN, DOUT, 256);
    gid_out_kernel<<<(NN * 4 + T - 1) / T, T, 0, stream>>>(zg16, Wg2, bg2, out);

    // ---- edge head ----
    gemm_edge_bn_kernel<<<(NP / 16 / 8) * (256 / 64), T, 0, stream>>>(
        h2f16, pairs, we1p, be1, e_gamma, e_beta, e_mean, e_var, ze16, 256);
    edge_out_kernel<<<(NP + T - 1) / T, T, 0, stream>>>(ze16, We2, be2, out);
}
